// TransformerSeq2Seq_16295105921355
// MI455X (gfx1250) — matrix operand order repaired and checked by reading, not compiled, not run on hardware
//
#include <hip/hip_runtime.h>
#include <math.h>

typedef _Float16 h16;
typedef __attribute__((ext_vector_type(16))) _Float16 v16h;
typedef __attribute__((ext_vector_type(8)))  _Float16 v8h;
typedef __attribute__((ext_vector_type(8)))  float    v8f;

static __device__ __forceinline__ v16h cat8(v8h lo, v8h hi) {
    return __builtin_shufflevector(lo, hi, 0, 1, 2, 3, 4, 5, 6, 7,
                                           8, 9, 10, 11, 12, 13, 14, 15);
}

// ---------------------------------------------------------------------------
// Batched, strided WMMA GEMM:  C[z] = act( A[z] @ B[z] + bias + resid )
//   A[z][m][k] = A  + z*sA + m*lda + k        (k contiguous, 16B aligned)
//   B[z][k][n] = Bm + z*sB + n*ldbN + k       (k contiguous, 32B aligned)
//   out[z][m][n] -> outF/outH + z*sO + m*ldo + n*ldon  (ldon lets V-proj
//   write V^T so every consumer GEMM sees a k-contiguous B operand)
// Each wave computes a 32x32 output block (2x2 tiles of 16x16);
// block = 4 waves stacked along M (128 rows / block). K % 64 == 0.
// ---------------------------------------------------------------------------
__global__ __launch_bounds__(128) void wmma_gemm_kernel(
    const h16* __restrict__ A, long long sA, int lda,
    const h16* __restrict__ Bm, long long sB, long long ldbN,
    const float* __restrict__ bias,
    const float* __restrict__ resid, long long sR, int ldr,
    float* __restrict__ outF, long long sOF, int ldof,
    h16* __restrict__ outH, long long sOH, int ldoh,
    int ldon, int M, int N, int K, int doGelu)
{
    const int z = blockIdx.z;
    const int lane = threadIdx.x & 31;
    const int wave = threadIdx.x >> 5;
    const int m0 = (blockIdx.y * 4 + wave) * 32;
    const int n0 = blockIdx.x * 32;
    const int lr = lane & 15;   // row (A) / col (B) / col (C) within tile
    const int hi = lane >> 4;   // half-wave select

    // A fragment per lane: halves {kb..kb+7, kb+16..kb+23}, kb = hi*8
    const h16* ap0 = A + (long long)z * sA + (long long)(m0 + lr) * lda + hi * 8;
    const h16* ap1 = ap0 + (long long)16 * lda;
    // B fragment per lane: halves {kb..kb+7, kb+16..kb+23}, kb = hi*8
    const h16* bp0 = Bm + (long long)z * sB + (long long)(n0 + lr) * ldbN + hi * 8;
    const h16* bp1 = bp0 + (long long)16 * ldbN;

    v8f acc00 = {}, acc01 = {}, acc10 = {}, acc11 = {};

    for (int k0 = 0; k0 < K; k0 += 64) {
        #pragma unroll
        for (int u = 0; u < 2; ++u) {
            const int k = k0 + u * 32;
            v16h a0 = cat8(*(const v8h*)(ap0 + k), *(const v8h*)(ap0 + k + 16));
            v16h a1 = cat8(*(const v8h*)(ap1 + k), *(const v8h*)(ap1 + k + 16));
            v16h b0 = cat8(*(const v8h*)(bp0 + k), *(const v8h*)(bp0 + k + 16));
            v16h b1 = cat8(*(const v8h*)(bp1 + k), *(const v8h*)(bp1 + k + 16));
            acc00 = __builtin_amdgcn_wmma_f32_16x16x32_f16(false, a0, false, b0, (short)0, acc00, false, false);
            acc01 = __builtin_amdgcn_wmma_f32_16x16x32_f16(false, a0, false, b1, (short)0, acc01, false, false);
            acc10 = __builtin_amdgcn_wmma_f32_16x16x32_f16(false, a1, false, b0, (short)0, acc10, false, false);
            acc11 = __builtin_amdgcn_wmma_f32_16x16x32_f16(false, a1, false, b1, (short)0, acc11, false, false);
        }
    }

    // C/D layout per tile: lane col n = lr; VGPR r -> row r + hi*8
    const int nA = n0 + lr;
    const int nB = nA + 16;
    const float bsA = bias ? bias[nA] : 0.0f;
    const float bsB = bias ? bias[nB] : 0.0f;

    auto emit = [&](float v, int m, int n, float bs) {
        v += bs;
        if (resid)  v += resid[(long long)z * sR + (long long)m * ldr + n];
        if (doGelu) v = 0.5f * v * (1.0f + erff(v * 0.70710678118654752f));
        const long long o = (long long)m * ldof * 0;  // (keep types simple)
        (void)o;
        if (outF) outF[(long long)z * sOF + (long long)m * ldof + (long long)n * ldon] = v;
        if (outH) outH[(long long)z * sOH + (long long)m * ldoh + (long long)n * ldon] = (h16)v;
    };

    #pragma unroll
    for (int r = 0; r < 8; ++r) {
        const int mA = m0 + r + hi * 8;
        const int mB = mA + 16;
        emit(acc00[r], mA, nA, bsA);
        emit(acc01[r], mA, nB, bsB);
        emit(acc10[r], mB, nA, bsA);
        emit(acc11[r], mB, nB, bsB);
    }
}

// ---------------------------------------------------------------------------
// LayerNorm: one row per block, f32 in -> f16 out (feeds WMMA A operand)
// ---------------------------------------------------------------------------
__global__ __launch_bounds__(128) void ln_kernel(
    const float* __restrict__ x, const float* __restrict__ g,
    const float* __restrict__ b, h16* __restrict__ out, int D)
{
    __shared__ float red[128];
    const long long row = blockIdx.x;
    const float* xr = x + row * D;
    const int tid = threadIdx.x;

    float s = 0.0f;
    for (int i = tid; i < D; i += 128) s += xr[i];
    red[tid] = s; __syncthreads();
    for (int o = 64; o > 0; o >>= 1) { if (tid < o) red[tid] += red[tid + o]; __syncthreads(); }
    const float mean = red[0] / (float)D;
    __syncthreads();

    float vs = 0.0f;
    for (int i = tid; i < D; i += 128) { float d = xr[i] - mean; vs += d * d; }
    red[tid] = vs; __syncthreads();
    for (int o = 64; o > 0; o >>= 1) { if (tid < o) red[tid] += red[tid + o]; __syncthreads(); }
    const float rstd = rsqrtf(red[0] / (float)D + 1e-5f);

    for (int i = tid; i < D; i += 128)
        out[row * D + i] = (h16)((xr[i] - mean) * rstd * g[i] + b[i]);
}

// ---------------------------------------------------------------------------
// Row softmax with scale + optional causal mask; f32 scores -> f16 probs
// grid = (Sq, H) for one batch; row = h*Sq + q
// ---------------------------------------------------------------------------
__global__ __launch_bounds__(128) void softmax_kernel(
    const float* __restrict__ sc, h16* __restrict__ pr,
    int Sk, float scale, int causal)
{
    __shared__ float red[128];
    const int q = blockIdx.x;
    const long long row = (long long)blockIdx.y * gridDim.x + q;
    const float* sr = sc + row * Sk;
    h16* orow = pr + row * Sk;
    const int tid = threadIdx.x;

    float mx = -3.0e38f;
    for (int i = tid; i < Sk; i += 128) {
        float v = sr[i] * scale;
        if (causal && i > q) v = -3.0e38f;
        mx = fmaxf(mx, v);
    }
    red[tid] = mx; __syncthreads();
    for (int o = 64; o > 0; o >>= 1) { if (tid < o) red[tid] = fmaxf(red[tid], red[tid + o]); __syncthreads(); }
    mx = red[0]; __syncthreads();

    float sm = 0.0f;
    for (int i = tid; i < Sk; i += 128) {
        float v = sr[i] * scale;
        if (causal && i > q) v = -3.0e38f;
        sm += __expf(v - mx);
    }
    red[tid] = sm; __syncthreads();
    for (int o = 64; o > 0; o >>= 1) { if (tid < o) red[tid] += red[tid + o]; __syncthreads(); }
    const float inv = 1.0f / red[0];

    for (int i = tid; i < Sk; i += 128) {
        float v = sr[i] * scale;
        if (causal && i > q) v = -3.0e38f;
        orow[i] = (h16)(__expf(v - mx) * inv);
    }
}

// token embedding gather + sinusoidal positional add
__global__ void embed_kernel(const int* __restrict__ ids,
                             const float* __restrict__ emb,
                             const float* __restrict__ pos,
                             float* __restrict__ out,
                             int S, int D, long long total)
{
    long long i = (long long)blockIdx.x * 256 + threadIdx.x;
    if (i >= total) return;
    long long t = i / D;
    int d = (int)(i % D);
    int s = (int)(t % S);
    out[i] = emb[(long long)ids[t] * D + d] + pos[(long long)s * D + d];
}

// convert f32 [z][K][N] weight -> f16 [z][N][K] (K-contiguous for B fragments)
__global__ void convtr_kernel(const float* __restrict__ src,
                              h16* __restrict__ dst, int K, int N)
{
    const long long per = (long long)K * N;
    const float* s = src + (long long)blockIdx.z * per;
    h16* d = dst + (long long)blockIdx.z * per;
    long long i = (long long)blockIdx.x * 256 + threadIdx.x;
    if (i < per) {
        int k = (int)(i / N), n = (int)(i % N);
        d[(long long)n * K + k] = (h16)s[i];
    }
}

__global__ void f2h_kernel(const float* __restrict__ src, h16* __restrict__ dst, long long n)
{
    long long i = (long long)blockIdx.x * 256 + threadIdx.x;
    if (i < n) dst[i] = (h16)src[i];
}

// ---------------------------------------------------------------------------
// Host orchestration
// ---------------------------------------------------------------------------
static inline void launch_gemm(hipStream_t st,
    const h16* A, long long sA, int lda,
    const h16* B, long long sB, long long ldbN,
    const float* bias,
    const float* resid, long long sR, int ldr,
    float* outF, long long sOF, int ldof,
    h16* outH, long long sOH, int ldoh,
    int ldon, int M, int N, int K, int gelu, int batches)
{
    dim3 grid(N / 32, M / 128, batches);
    wmma_gemm_kernel<<<grid, dim3(128), 0, st>>>(
        A, sA, lda, B, sB, ldbN, bias, resid, sR, ldr,
        outF, sOF, ldof, outH, sOH, ldoh, ldon, M, N, K, gelu);
}

extern "C" void kernel_launch(void* const* d_in, const int* in_sizes, int n_in,
                              void* d_out, int out_size, void* d_ws, size_t ws_size,
                              hipStream_t stream)
{
    (void)in_sizes; (void)n_in; (void)out_size; (void)ws_size;
    const int Lc = 6, D = 512, H = 8, F = 2048, V = 8000, B = 8, S = 512, HDm = 64;
    const int M = B * S;            // 4096 tokens
    const float scale = 0.125f;     // HD^-0.5

    const int*   src_ids   = (const int*)d_in[0];
    const int*   tgt_ids   = (const int*)d_in[1];
    const float* src_emb   = (const float*)d_in[2];
    const float* tgt_emb   = (const float*)d_in[3];
    const float* pos_emb   = (const float*)d_in[4];
    const float* enc_attn_w= (const float*)d_in[5];
    const float* enc_attn_b= (const float*)d_in[6];
    const float* enc_ln_g  = (const float*)d_in[7];
    const float* enc_ln_b  = (const float*)d_in[8];
    const float* enc_ff1_w = (const float*)d_in[9];
    const float* enc_ff1_b = (const float*)d_in[10];
    const float* enc_ff2_w = (const float*)d_in[11];
    const float* enc_ff2_b = (const float*)d_in[12];
    const float* dec_self_w= (const float*)d_in[13];
    const float* dec_self_b= (const float*)d_in[14];
    const float* dec_cross_w=(const float*)d_in[15];
    const float* dec_cross_b=(const float*)d_in[16];
    const float* dec_ln_g  = (const float*)d_in[17];
    const float* dec_ln_b  = (const float*)d_in[18];
    const float* dec_ff1_w = (const float*)d_in[19];
    const float* dec_ff1_b = (const float*)d_in[20];
    const float* dec_ff2_w = (const float*)d_in[21];
    const float* dec_ff2_b = (const float*)d_in[22];
    const float* lm_w      = (const float*)d_in[23];
    const float* lm_b      = (const float*)d_in[24];
    float* logits = (float*)d_out;

    // ---- workspace carve-up (256B aligned) ----
    char* w = (char*)d_ws;
    size_t off = 0;
    auto alc = [&](size_t bytes) { size_t o = off; off += (bytes + 255) & ~(size_t)255; return o; };
    h16* encA_t  = (h16*)(w + alc((size_t)Lc * 4 * D * D * 2));
    h16* encF1_t = (h16*)(w + alc((size_t)Lc * D * F * 2));
    h16* encF2_t = (h16*)(w + alc((size_t)Lc * F * D * 2));
    h16* decS_t  = (h16*)(w + alc((size_t)Lc * 4 * D * D * 2));
    h16* decC_t  = (h16*)(w + alc((size_t)Lc * 4 * D * D * 2));
    h16* decF1_t = (h16*)(w + alc((size_t)Lc * D * F * 2));
    h16* decF2_t = (h16*)(w + alc((size_t)Lc * F * D * 2));
    h16* lm_t    = (h16*)(w + alc((size_t)D * V * 2));
    float* hbuf  = (float*)(w + alc((size_t)M * D * 4));   // residual stream (enc then dec)
    h16* memh    = (h16*)(w + alc((size_t)M * D * 2));     // encoder memory, f16
    h16* hn      = (h16*)(w + alc((size_t)M * D * 2));     // LN output
    h16* qh      = (h16*)(w + alc((size_t)M * D * 2));
    h16* kh      = (h16*)(w + alc((size_t)M * D * 2));
    h16* vt      = (h16*)(w + alc((size_t)M * D * 2));     // V^T: [B][D][S]
    h16* ao      = (h16*)(w + alc((size_t)M * D * 2));     // attention context
    h16* ffh     = (h16*)(w + alc((size_t)M * F * 2));     // FFN hidden
    float* scores= (float*)(w + alc((size_t)H * S * S * 4)); // per-batch scores
    h16* probs   = (h16*)(w + alc((size_t)H * S * S * 2));   // per-batch probs
    h16* yh      = (h16*)(w + alc((size_t)M * D * 2));     // LM-head input

    // ---- weight convert + transpose to f16 [N][K] ----
    auto ctr = [&](const float* s, h16* d, int K, int N, int z) {
        long long per = (long long)K * N;
        dim3 g((unsigned)((per + 255) / 256), 1, (unsigned)z);
        convtr_kernel<<<g, dim3(256), 0, stream>>>(s, d, K, N);
    };
    ctr(enc_attn_w, encA_t, D, D, Lc * 4);
    ctr(enc_ff1_w,  encF1_t, D, F, Lc);
    ctr(enc_ff2_w,  encF2_t, F, D, Lc);
    ctr(dec_self_w, decS_t, D, D, Lc * 4);
    ctr(dec_cross_w,decC_t, D, D, Lc * 4);
    ctr(dec_ff1_w,  decF1_t, D, F, Lc);
    ctr(dec_ff2_w,  decF2_t, F, D, Lc);
    ctr(lm_w,       lm_t,   D, V, 1);

    const long long tot = (long long)M * D;

    // q,k projections over all tokens; v projection per batch writing V^T
    auto qkv = [&](const h16* src, const h16* wt, const float* bias) {
        launch_gemm(stream, src, 0, D, wt + 0 * (size_t)D * D, 0, D,
                    bias + 0 * D, nullptr, 0, 0, nullptr, 0, 0,
                    qh, 0, D, 1, M, D, D, 0, 1);
        launch_gemm(stream, src, 0, D, wt + 1 * (size_t)D * D, 0, D,
                    bias + 1 * D, nullptr, 0, 0, nullptr, 0, 0,
                    kh, 0, D, 1, M, D, D, 0, 1);
        for (int b = 0; b < B; ++b)
            launch_gemm(stream, src + (size_t)b * S * D, 0, D, wt + 2 * (size_t)D * D, 0, D,
                        bias + 2 * D, nullptr, 0, 0, nullptr, 0, 0,
                        vt + (size_t)b * D * S, 0, /*ldoh(m)=*/1, /*ldon(n)=*/S,
                        S, D, D, 0, 1);
    };

    auto attention = [&](int causal) {
        for (int b = 0; b < B; ++b) {
            const long long boff = (long long)b * S * D;
            // scores[h] = Q_bh @ K_bh^T   (K^T view: n-stride D, k contiguous)
            launch_gemm(stream, qh + boff, HDm, D, kh + boff, HDm, D,
                        nullptr, nullptr, 0, 0,
                        scores, (long long)S * S, S, nullptr, 0, 0,
                        1, S, S, HDm, 0, H);
            softmax_kernel<<<dim3(S, H), dim3(128), 0, stream>>>(scores, probs, S, scale, causal);
            // ctx[h] = P @ V_bh  (V^T view: n-stride S, k contiguous)
            launch_gemm(stream, probs, (long long)S * S, S,
                        vt + (size_t)b * D * S, (long long)HDm * S, S,
                        nullptr, nullptr, 0, 0,
                        nullptr, 0, 0, ao + boff, HDm, D,
                        1, S, HDm, S, 0, H);
        }
    };

    // ================= encoder =================
    embed_kernel<<<dim3((unsigned)((tot + 255) / 256)), dim3(256), 0, stream>>>(
        src_ids, src_emb, pos_emb, hbuf, S, D, tot);
    for (int i = 0; i < Lc; ++i) {
        ln_kernel<<<dim3(M), dim3(128), 0, stream>>>(hbuf, enc_ln_g + (i*2+0)*D, enc_ln_b + (i*2+0)*D, hn, D);
        qkv(hn, encA_t + (size_t)(i*4)*D*D, enc_attn_b + (size_t)(i*4)*D);
        attention(0);
        launch_gemm(stream, ao, 0, D, encA_t + (size_t)(i*4+3)*D*D, 0, D,
                    enc_attn_b + (i*4+3)*D, hbuf, 0, D,
                    hbuf, 0, D, nullptr, 0, 0, 1, M, D, D, 0, 1);
        ln_kernel<<<dim3(M), dim3(128), 0, stream>>>(hbuf, enc_ln_g + (i*2+1)*D, enc_ln_b + (i*2+1)*D, hn, D);
        launch_gemm(stream, hn, 0, D, encF1_t + (size_t)i*D*F, 0, D,
                    enc_ff1_b + (size_t)i*F, nullptr, 0, 0,
                    nullptr, 0, 0, ffh, 0, F, 1, M, F, D, 1, 1);
        launch_gemm(stream, ffh, 0, F, encF2_t + (size_t)i*D*F, 0, F,
                    enc_ff2_b + (size_t)i*D, hbuf, 0, D,
                    hbuf, 0, D, nullptr, 0, 0, 1, M, D, F, 0, 1);
    }
    f2h_kernel<<<dim3((unsigned)((tot + 255) / 256)), dim3(256), 0, stream>>>(hbuf, memh, tot);

    // ================= decoder =================
    embed_kernel<<<dim3((unsigned)((tot + 255) / 256)), dim3(256), 0, stream>>>(
        tgt_ids, tgt_emb, pos_emb, hbuf, S, D, tot);
    for (int i = 0; i < Lc; ++i) {
        // -- causal self-attention --
        ln_kernel<<<dim3(M), dim3(128), 0, stream>>>(hbuf, dec_ln_g + (i*3+0)*D, dec_ln_b + (i*3+0)*D, hn, D);
        qkv(hn, decS_t + (size_t)(i*4)*D*D, dec_self_b + (size_t)(i*4)*D);
        attention(1);
        launch_gemm(stream, ao, 0, D, decS_t + (size_t)(i*4+3)*D*D, 0, D,
                    dec_self_b + (i*4+3)*D, hbuf, 0, D,
                    hbuf, 0, D, nullptr, 0, 0, 1, M, D, D, 0, 1);
        // -- cross-attention (K,V from encoder memory) --
        ln_kernel<<<dim3(M), dim3(128), 0, stream>>>(hbuf, dec_ln_g + (i*3+1)*D, dec_ln_b + (i*3+1)*D, hn, D);
        launch_gemm(stream, hn, 0, D, decC_t + (size_t)(i*4+0)*D*D, 0, D,
                    dec_cross_b + (i*4+0)*D, nullptr, 0, 0,
                    nullptr, 0, 0, qh, 0, D, 1, M, D, D, 0, 1);
        launch_gemm(stream, memh, 0, D, decC_t + (size_t)(i*4+1)*D*D, 0, D,
                    dec_cross_b + (i*4+1)*D, nullptr, 0, 0,
                    nullptr, 0, 0, kh, 0, D, 1, M, D, D, 0, 1);
        for (int b = 0; b < B; ++b)
            launch_gemm(stream, memh + (size_t)b * S * D, 0, D,
                        decC_t + (size_t)(i*4+2)*D*D, 0, D,
                        dec_cross_b + (i*4+2)*D, nullptr, 0, 0,
                        nullptr, 0, 0, vt + (size_t)b * D * S, 0, 1, S,
                        S, D, D, 0, 1);
        attention(0);
        launch_gemm(stream, ao, 0, D, decC_t + (size_t)(i*4+3)*D*D, 0, D,
                    dec_cross_b + (i*4+3)*D, hbuf, 0, D,
                    hbuf, 0, D, nullptr, 0, 0, 1, M, D, D, 0, 1);
        // -- FFN --
        ln_kernel<<<dim3(M), dim3(128), 0, stream>>>(hbuf, dec_ln_g + (i*3+2)*D, dec_ln_b + (i*3+2)*D, hn, D);
        launch_gemm(stream, hn, 0, D, decF1_t + (size_t)i*D*F, 0, D,
                    dec_ff1_b + (size_t)i*F, nullptr, 0, 0,
                    nullptr, 0, 0, ffh, 0, F, 1, M, F, D, 1, 1);
        launch_gemm(stream, ffh, 0, F, decF2_t + (size_t)i*D*F, 0, F,
                    dec_ff2_b + (size_t)i*D, hbuf, 0, D,
                    hbuf, 0, D, nullptr, 0, 0, 1, M, D, F, 0, 1);
    }

    // ================= LM head =================
    f2h_kernel<<<dim3((unsigned)((tot + 255) / 256)), dim3(256), 0, stream>>>(hbuf, yh, tot);
    launch_gemm(stream, yh, 0, D, lm_t, 0, D,
                lm_b, nullptr, 0, 0,
                logits, 0, V, nullptr, 0, 0, 1, M, V, D, 0, 1);
}
